// SAGE_87376814670104
// MI455X (gfx1250) — compile-verified, matrix operand order checked by reading, not run on hardware
//
#include <hip/hip_runtime.h>

#define HIDC 128
#define WAVES_PER_BLOCK 8

typedef __attribute__((ext_vector_type(16))) __bf16 v16bf;
typedef __attribute__((ext_vector_type(8)))  float  v8f;

// ISA 7.12.2: 16-bit A (16x32) / B (32x16) fragment K-offset for element e
// (e = 2*vgpr + pos), given which K-half (lane>>4) this lane covers.
__device__ __forceinline__ int wmma_koff(int e, int half) {
    int v = e >> 1, p = e & 1;
    return (v < 4) ? (v * 2 + half * 8 + p)
                   : (16 + (v - 4) * 2 + half * 8 + p);
}

// A fragment: lane covers one row (l&15). koff is two contiguous 8-element
// runs: [kb + half*8, +8) and [kb + 16 + half*8, +8). Load 4x float4, convert.
__device__ __forceinline__ v16bf load_a_frag(const float* __restrict__ rowp,
                                             float s, int kb, int half) {
    const float* p0 = rowp + kb + half * 8;
    float4 x0 = *(const float4*)(p0);
    float4 x1 = *(const float4*)(p0 + 4);
    float4 x2 = *(const float4*)(p0 + 16);
    float4 x3 = *(const float4*)(p0 + 20);
    v16bf f;
    f[0]  = (__bf16)(x0.x * s); f[1]  = (__bf16)(x0.y * s);
    f[2]  = (__bf16)(x0.z * s); f[3]  = (__bf16)(x0.w * s);
    f[4]  = (__bf16)(x1.x * s); f[5]  = (__bf16)(x1.y * s);
    f[6]  = (__bf16)(x1.z * s); f[7]  = (__bf16)(x1.w * s);
    f[8]  = (__bf16)(x2.x * s); f[9]  = (__bf16)(x2.y * s);
    f[10] = (__bf16)(x2.z * s); f[11] = (__bf16)(x2.w * s);
    f[12] = (__bf16)(x3.x * s); f[13] = (__bf16)(x3.y * s);
    f[14] = (__bf16)(x3.z * s); f[15] = (__bf16)(x3.w * s);
    return f;
}

// Stage weight matrix W (fp32, [k][n] row-major, 128 x ncols) into LDS in
// *fragment order*: slot f = (ks*NT + nt)*32 + lane holds that fragment's 16
// bf16 elements contiguously (32B aligned). Inner loop then reads one
// fragment with a single 32-byte LDS access (2x ds_load_b128).
__device__ __forceinline__ void stage_w_frags(__bf16* __restrict__ lw,
                                              const float* __restrict__ W,
                                              int ncols, int tid) {
    const int NT = ncols >> 4;
    const int nfrag = 4 * NT * 32;
    for (int f = tid; f < nfrag; f += 256) {
        const int lane_f = f & 31;
        const int slot   = f >> 5;
        const int nt     = slot % NT;
        const int ks     = slot / NT;
        const int half   = lane_f >> 4;
        const int n      = nt * 16 + (lane_f & 15);
        __bf16* d = lw + (size_t)f * 16;
#pragma unroll
        for (int e = 0; e < 16; ++e) {
            int k = ks * 32 + wmma_koff(e, half);
            d[e] = (__bf16)W[k * ncols + n];
        }
    }
}

// out[m][n] = (A1[m][:] * s1[m]) @ W1 + (A2 @ W2)? + bias[n] + resid[m][n]?
// K fixed = 128. ncols = 16*NT (128 or 64). One wave per 16-row tile.
__global__ __launch_bounds__(256) void wmma_gemm_kernel(
    const float* __restrict__ A1, const float* __restrict__ scale1,
    const float* __restrict__ W1,
    const float* __restrict__ A2, const float* __restrict__ W2,
    const float* __restrict__ bias, const float* __restrict__ resid,
    float* __restrict__ out, int ntiles, int ncols) {
    __shared__ __bf16 lw1[HIDC * HIDC];   // 32KB, fragment-ordered
    __shared__ __bf16 lw2[HIDC * HIDC];   // 32KB, fragment-ordered

    const int tid = threadIdx.x;
    stage_w_frags(lw1, W1, ncols, tid);
    if (A2) stage_w_frags(lw2, W2, ncols, tid);
    __syncthreads();

    const int wave = tid >> 5;
    const int lane = tid & 31;
    const int tile = blockIdx.x * WAVES_PER_BLOCK + wave;
    if (tile >= ntiles) return;                 // wave-uniform: EXEC stays full

    const int half = lane >> 4;
    const int id   = lane & 15;
    const int row  = tile * 16 + id;
    const int NT   = ncols >> 4;

    // per-lane fragment base inside LDS (elements)
    const __bf16* lf1 = lw1 + lane * 16;
    const __bf16* lf2 = lw2 + lane * 16;

    const float s1 = scale1 ? scale1[row] : 1.0f;
    const float* a1p = A1 + (size_t)row * HIDC;
    v16bf a1[4];
#pragma unroll
    for (int ks = 0; ks < 4; ++ks) a1[ks] = load_a_frag(a1p, s1, ks * 32, half);

    v16bf a2[4];
    if (A2) {
        const float* a2p = A2 + (size_t)row * HIDC;
#pragma unroll
        for (int ks = 0; ks < 4; ++ks) a2[ks] = load_a_frag(a2p, 1.0f, ks * 32, half);
    }

    for (int nt = 0; nt < NT; ++nt) {
        const int col = nt * 16 + id;
        const float bv = bias ? bias[col] : 0.0f;
        v8f accA, accB;
#pragma unroll
        for (int v = 0; v < 8; ++v) { accA[v] = bv; accB[v] = 0.0f; }

        if (A2) {
            // Two independent accumulator chains interleaved per k-step (no
            // WMMA->WMMA RAW, ISA 7.12.1) + B-fragments prefetched one step
            // ahead so ds_load_b128s overlap the current WMMAs.
            v16bf b1 = *(const v16bf*)(lf1 + (size_t)nt * 512);
            v16bf b2 = *(const v16bf*)(lf2 + (size_t)nt * 512);
#pragma unroll
            for (int ks = 0; ks < 4; ++ks) {
                v16bf n1, n2;
                if (ks < 3) {
                    n1 = *(const v16bf*)(lf1 + (size_t)((ks + 1) * NT + nt) * 512);
                    n2 = *(const v16bf*)(lf2 + (size_t)((ks + 1) * NT + nt) * 512);
                }
                accA = __builtin_amdgcn_wmma_f32_16x16x32_bf16(
                    false, a1[ks], false, b1, (short)0, accA, false, false);
                accB = __builtin_amdgcn_wmma_f32_16x16x32_bf16(
                    false, a2[ks], false, b2, (short)0, accB, false, false);
                b1 = n1; b2 = n2;
            }
        } else {
            // single matrix: split into even/odd k-step chains to avoid the
            // WMMA->WMMA accumulator RAW hazard NOPs
            v16bf b0 = *(const v16bf*)(lf1 + (size_t)(0 * NT + nt) * 512);
            v16bf b1 = *(const v16bf*)(lf1 + (size_t)(1 * NT + nt) * 512);
            v16bf b2 = *(const v16bf*)(lf1 + (size_t)(2 * NT + nt) * 512);
            v16bf b3 = *(const v16bf*)(lf1 + (size_t)(3 * NT + nt) * 512);
            accA = __builtin_amdgcn_wmma_f32_16x16x32_bf16(
                false, a1[0], false, b0, (short)0, accA, false, false);
            accB = __builtin_amdgcn_wmma_f32_16x16x32_bf16(
                false, a1[1], false, b1, (short)0, accB, false, false);
            accA = __builtin_amdgcn_wmma_f32_16x16x32_bf16(
                false, a1[2], false, b2, (short)0, accA, false, false);
            accB = __builtin_amdgcn_wmma_f32_16x16x32_bf16(
                false, a1[3], false, b3, (short)0, accB, false, false);
        }

        // D layout: element v -> row (v + 8*half), col id  (ISA 7.12.2)
#pragma unroll
        for (int v = 0; v < 8; ++v) {
            const int m = tile * 16 + v + 8 * half;
            const size_t off = (size_t)m * ncols + col;
            float r = accB[v];
            if (resid) r += resid[off];
            out[off] = accA[v] + r;
        }
    }
}

// degree accumulation: one thread per edge
__global__ __launch_bounds__(256) void deg_kernel(const int* __restrict__ dst,
                                                  float* __restrict__ deg, int E) {
    int e = blockIdx.x * 256 + threadIdx.x;
    if (e < E) atomicAdd(&deg[dst[e]], 1.0f);
}

__global__ __launch_bounds__(256) void rdeg_kernel(float* __restrict__ deg, int n) {
    int i = blockIdx.x * 256 + threadIdx.x;
    if (i < n) deg[i] = 1.0f / fmaxf(deg[i], 1.0f);
}

// scatter-add: one wave per edge (32 lanes x float4 = 128 features)
__global__ __launch_bounds__(256) void scatter_kernel(
    const float* __restrict__ h, const int* __restrict__ src,
    const int* __restrict__ dst, float* __restrict__ accum, int E) {
    long long idx = (long long)blockIdx.x * 256 + threadIdx.x;
    int e = (int)(idx >> 5);
    if (e >= E) return;
    int f = ((int)idx & 31) * 4;
    int s = src[e], d = dst[e];
    float4 v = *(const float4*)(h + (size_t)s * HIDC + f);
    float* ap = accum + (size_t)d * HIDC + f;
    atomicAdd(ap + 0, v.x);
    atomicAdd(ap + 1, v.y);
    atomicAdd(ap + 2, v.z);
    atomicAdd(ap + 3, v.w);
}

// LayerNorm + ReLU over 128 features: one wave per row
__global__ __launch_bounds__(256) void ln_relu_kernel(
    const float* __restrict__ x, const float* __restrict__ gamma,
    const float* __restrict__ beta, float* __restrict__ out, int nrows) {
    int wave = threadIdx.x >> 5, lane = threadIdx.x & 31;
    int row = blockIdx.x * WAVES_PER_BLOCK + wave;
    if (row >= nrows) return;
    int f = lane * 4;
    float4 v = *(const float4*)(x + (size_t)row * HIDC + f);
    float s = v.x + v.y + v.z + v.w;
#pragma unroll
    for (int o = 16; o > 0; o >>= 1) s += __shfl_xor(s, o, 32);
    float mu = s * (1.0f / 128.0f);
    float dx = v.x - mu, dy = v.y - mu, dz = v.z - mu, dw = v.w - mu;
    float ss = dx * dx + dy * dy + dz * dz + dw * dw;
#pragma unroll
    for (int o = 16; o > 0; o >>= 1) ss += __shfl_xor(ss, o, 32);
    float inv = rsqrtf(ss * (1.0f / 128.0f) + 1e-5f);
    float4 g = *(const float4*)(gamma + f);
    float4 b = *(const float4*)(beta + f);
    float4 o4;
    o4.x = fmaxf(dx * inv * g.x + b.x, 0.0f);
    o4.y = fmaxf(dy * inv * g.y + b.y, 0.0f);
    o4.z = fmaxf(dz * inv * g.z + b.z, 0.0f);
    o4.w = fmaxf(dw * inv * g.w + b.w, 0.0f);
    *(float4*)(out + (size_t)row * HIDC + f) = o4;
}

extern "C" void kernel_launch(void* const* d_in, const int* in_sizes, int n_in,
                              void* d_out, int out_size, void* d_ws, size_t ws_size,
                              hipStream_t stream) {
    const float* x     = (const float*)d_in[0];
    const int*   ei    = (const int*)d_in[1];
    // d_in[2] = edge_attr: unused by the reference
    const float* W_ne  = (const float*)d_in[3];
    const float* b_ne  = (const float*)d_in[4];
    const float* W_l   = (const float*)d_in[5];
    const float* b_l   = (const float*)d_in[6];
    const float* W_r   = (const float*)d_in[7];
    const float* gamma = (const float*)d_in[8];
    const float* beta  = (const float*)d_in[9];
    const float* W_lin = (const float*)d_in[10];
    const float* b_lin = (const float*)d_in[11];

    const int N = in_sizes[0] / HIDC;
    const int E = in_sizes[1] / 2;
    const int* src = ei;
    const int* dst = ei + E;

    const size_t nf = (size_t)N * HIDC;
    float* buf0 = (float*)d_ws;     // h0, later r
    float* buf1 = buf0 + nf;        // scatter accumulator
    float* buf2 = buf1 + nf;        // h1 (h2 written in place with residual)
    float* buf3 = buf2 + nf;        // o
    float* deg  = buf3 + nf;        // degree -> reciprocal degree

    const int ntiles = (N + 15) / 16;                       // 3125
    dim3 gemm_grid((ntiles + WAVES_PER_BLOCK - 1) / WAVES_PER_BLOCK);
    dim3 ln_grid((N + WAVES_PER_BLOCK - 1) / WAVES_PER_BLOCK);
    int deg_blocks = (E + 255) / 256;
    int sc_blocks  = (int)(((long long)E * 32 + 255) / 256);

    // reciprocal mean-degree (same dst both convs -> compute once per call)
    hipMemsetAsync(deg, 0, (size_t)N * sizeof(float), stream);
    deg_kernel<<<deg_blocks, 256, 0, stream>>>(dst, deg, E);
    rdeg_kernel<<<(N + 255) / 256, 256, 0, stream>>>(deg, N);

    // h0 = x @ W_ne + b_ne
    wmma_gemm_kernel<<<gemm_grid, 256, 0, stream>>>(
        x, nullptr, W_ne, nullptr, nullptr, b_ne, nullptr, buf0, ntiles, HIDC);

    // mean-aggregate h0
    hipMemsetAsync(buf1, 0, nf * sizeof(float), stream);
    scatter_kernel<<<sc_blocks, 256, 0, stream>>>(buf0, src, dst, buf1, E);

    // h1 = mean @ W_l + b_l + h0 @ W_r
    wmma_gemm_kernel<<<gemm_grid, 256, 0, stream>>>(
        buf1, deg, W_l, buf0, W_r, b_l, nullptr, buf2, ntiles, HIDC);

    // r = relu(LN(h1))
    ln_relu_kernel<<<ln_grid, 256, 0, stream>>>(buf2, gamma, beta, buf0, N);

    // mean-aggregate r
    hipMemsetAsync(buf1, 0, nf * sizeof(float), stream);
    scatter_kernel<<<sc_blocks, 256, 0, stream>>>(buf0, src, dst, buf1, E);

    // h2 = mean @ W_l + b_l + r @ W_r + h1   (residual fused, in place)
    wmma_gemm_kernel<<<gemm_grid, 256, 0, stream>>>(
        buf1, deg, W_l, buf0, W_r, b_l, buf2, buf2, ntiles, HIDC);

    // o = relu(LN(h2))
    ln_relu_kernel<<<ln_grid, 256, 0, stream>>>(buf2, gamma, beta, buf3, N);

    // out = o @ W_lin + b_lin   (50000 x 64 fp32)
    wmma_gemm_kernel<<<gemm_grid, 256, 0, stream>>>(
        buf3, nullptr, W_lin, nullptr, nullptr, b_lin, nullptr,
        (float*)d_out, ntiles, 64);
}
